// HungarianMatcher_56581899157958
// MI455X (gfx1250) — matrix operand order repaired and checked by reading, not compiled, not run on hardware
//
#include <hip/hip_runtime.h>
#include <hip/hip_bf16.h>
#include <math.h>

typedef __attribute__((ext_vector_type(16))) _Float16 v16h;
typedef __attribute__((ext_vector_type(8)))  _Float16 v8h;
typedef __attribute__((ext_vector_type(8)))  float    v8f;

#define NQ     400      // bs*Q
#define KCLS   134
#define MT     80
#define PFULL  65536    // H*W
#define PN     12544    // NUM_POINTS (= 392*32 = 49*256)
#define NCHK   49       // PN / 256 gather chunks
#define KSPLIT 14       // K-splits for the WMMA GEMM
#define KCHUNK (PN / KSPLIT)   // 896 = 28*32  (even # of K=32 steps)

// ------------------------------------------------- gather x -> f16 X, S + block partials
// grid (NQ, NCHK), block 256.  Partials: sp_part/sg_part [NQ][NCHK]
__global__ void hm_gather_x(const float* __restrict__ pm,
                            const int*   __restrict__ pidx,
                            _Float16* __restrict__ Xh, _Float16* __restrict__ Sh,
                            float* __restrict__ sp_part, float* __restrict__ sg_part) {
  int n = blockIdx.x;
  int j = blockIdx.y * 256 + threadIdx.x;
  int p = pidx[j];
  float x  = pm[(size_t)n * PFULL + p];
  float s  = 1.0f / (1.0f + __expf(-x));
  float sp = fmaxf(x, 0.0f) + log1pf(__expf(-fabsf(x)));   // softplus, stable
  Xh[(size_t)n * PN + j] = (_Float16)x;
  Sh[(size_t)n * PN + j] = (_Float16)s;

  __shared__ float rs[256], rp[256];
  rs[threadIdx.x] = s; rp[threadIdx.x] = sp;
  __syncthreads();
  for (int o = 128; o > 0; o >>= 1) {
    if (threadIdx.x < o) {
      rs[threadIdx.x] += rs[threadIdx.x + o];
      rp[threadIdx.x] += rp[threadIdx.x + o];
    }
    __syncthreads();
  }
  if (threadIdx.x == 0) {
    sg_part[n * NCHK + blockIdx.y] = rs[0];
    sp_part[n * NCHK + blockIdx.y] = rp[0];
  }
}

// ---------------------------------------- gather y -> transposed f16 Yt + block partials
// grid (MT, NCHK), block 256.  Yt layout: [PN][MT]
__global__ void hm_gather_y(const float* __restrict__ tm,
                            const int*   __restrict__ pidx,
                            _Float16* __restrict__ Yt, float* __restrict__ ys_part) {
  int m = blockIdx.x;
  int j = blockIdx.y * 256 + threadIdx.x;
  int p = pidx[j];
  float y = tm[(size_t)m * PFULL + p];
  Yt[(size_t)j * MT + m] = (_Float16)y;

  __shared__ float ry[256];
  ry[threadIdx.x] = y;
  __syncthreads();
  for (int o = 128; o > 0; o >>= 1) {
    if (threadIdx.x < o) ry[threadIdx.x] += ry[threadIdx.x + o];
    __syncthreads();
  }
  if (threadIdx.x == 0) ys_part[m * NCHK + blockIdx.y] = ry[0];
}

// -------------------------------- deterministic row reductions (no atomics)
// grid (NQ + MT) blocks of 32: rows 0..NQ-1 reduce sp/sg; rows NQ.. reduce ysum
__global__ void __launch_bounds__(32)
hm_reduce_rows(const float* __restrict__ sp_part, const float* __restrict__ sg_part,
               const float* __restrict__ ys_part,
               float* __restrict__ sp_sum, float* __restrict__ sg_sum,
               float* __restrict__ ysum) {
  int row = blockIdx.x;
  int lane = threadIdx.x;
  if (row < NQ) {
    float a = 0.0f, b = 0.0f;
    for (int c = lane; c < NCHK; c += 32) {
      a += sp_part[row * NCHK + c];
      b += sg_part[row * NCHK + c];
    }
    for (int o = 16; o > 0; o >>= 1) {
      a += __shfl_xor(a, o);
      b += __shfl_xor(b, o);
    }
    if (lane == 0) { sp_sum[row] = a; sg_sum[row] = b; }
  } else {
    int m = row - NQ;
    float a = 0.0f;
    for (int c = lane; c < NCHK; c += 32) a += ys_part[m * NCHK + c];
    for (int o = 16; o > 0; o >>= 1) a += __shfl_xor(a, o);
    if (lane == 0) ysum[m] = a;
  }
}

// ------------------------------------------------- dual WMMA GEMM, K-split
// grid (25, 5, KSPLIT), one wave per (16x16 tile, K-chunk).
// XYp[ks][n][m] = sum_{p in chunk} x[n,p]*y[m,p] ; SYp likewise with sigmoid(x)
__global__ void __launch_bounds__(32)
hm_wmma(const _Float16* __restrict__ Xh, const _Float16* __restrict__ Sh,
        const _Float16* __restrict__ Yt,
        float* __restrict__ XYp, float* __restrict__ SYp) {
  int tn = blockIdx.x, tm = blockIdx.y, ks = blockIdx.z;
  int lane = threadIdx.x;
  int r = lane & 15, hf = lane >> 4;

  const int kbeg = ks * KCHUNK;

  // A layout (16-bit A 16x32): lane (r,hf) holds K = {k0+hf*8..+8} U {k0+16+hf*8..+8}
  const _Float16* pax = Xh + (size_t)(tn * 16 + r) * PN + kbeg + hf * 8;
  const _Float16* pas = Sh + (size_t)(tn * 16 + r) * PN + kbeg + hf * 8;
  // B layout (32x16): lane (r,hf) holds column N=r, K = {k0+hf*8..+8} U {k0+16+hf*8..+8}
  const _Float16* pb  = Yt + (size_t)kbeg * MT + tm * 16 + r;

  v8f accx = {};
  v8f accs = {};

  #pragma unroll 1
  for (int it = 0; it < KCHUNK / 64; ++it) {   // two K=32 steps per iteration
    v16h B0, B1;
    #pragma unroll
    for (int s = 0; s < 16; ++s) {
      int kk = (s < 8) ? (hf * 8 + s) : (16 + hf * 8 + (s - 8));
      B0[s] = pb[(size_t)kk * MT];
      B1[s] = pb[(size_t)(kk + 32) * MT];
    }

    v8h xl0 = *(const v8h*)(pax);
    v8h xh0 = *(const v8h*)(pax + 16);
    v8h xl1 = *(const v8h*)(pax + 32);
    v8h xh1 = *(const v8h*)(pax + 48);
    v8h sl0 = *(const v8h*)(pas);
    v8h sh0 = *(const v8h*)(pas + 16);
    v8h sl1 = *(const v8h*)(pas + 32);
    v8h sh1 = *(const v8h*)(pas + 48);

    v16h Ax0 = __builtin_shufflevector(xl0, xh0, 0,1,2,3,4,5,6,7,8,9,10,11,12,13,14,15);
    v16h Ax1 = __builtin_shufflevector(xl1, xh1, 0,1,2,3,4,5,6,7,8,9,10,11,12,13,14,15);
    v16h As0 = __builtin_shufflevector(sl0, sh0, 0,1,2,3,4,5,6,7,8,9,10,11,12,13,14,15);
    v16h As1 = __builtin_shufflevector(sl1, sh1, 0,1,2,3,4,5,6,7,8,9,10,11,12,13,14,15);

    accx = __builtin_amdgcn_wmma_f32_16x16x32_f16(
        false, Ax0, false, B0, (short)0, accx, false, false);
    accs = __builtin_amdgcn_wmma_f32_16x16x32_f16(
        false, As0, false, B0, (short)0, accs, false, false);
    accx = __builtin_amdgcn_wmma_f32_16x16x32_f16(
        false, Ax1, false, B1, (short)0, accx, false, false);
    accs = __builtin_amdgcn_wmma_f32_16x16x32_f16(
        false, As1, false, B1, (short)0, accs, false, false);

    pax += 64;
    pas += 64;
    pb  += (size_t)64 * MT;
  }

  // D layout: VGPR v, lane -> row = v + hf*8, col = r
  float* px = XYp + (size_t)ks * NQ * MT + (size_t)(tn * 16 + hf * 8) * MT + tm * 16 + r;
  float* ps = SYp + (size_t)ks * NQ * MT + (size_t)(tn * 16 + hf * 8) * MT + tm * 16 + r;
  #pragma unroll
  for (int v = 0; v < 8; ++v) {
    px[(size_t)v * MT] = accx[v];
    ps[(size_t)v * MT] = accs[v];
  }
}

// --------------------------------- sum the KSPLIT partial tiles (deterministic)
// grid (NQ*MT/256), block 256
__global__ void hm_reduce_tiles(const float* __restrict__ XYp,
                                const float* __restrict__ SYp,
                                float* __restrict__ XY, float* __restrict__ SY) {
  int i = blockIdx.x * 256 + threadIdx.x;
  float ax = 0.0f, asum = 0.0f;
  #pragma unroll
  for (int s = 0; s < KSPLIT; ++s) {
    ax   += XYp[(size_t)s * NQ * MT + i];
    asum += SYp[(size_t)s * NQ * MT + i];
  }
  XY[i] = ax;
  SY[i] = asum;
}

// ----------------------------------------------------- finalize: one wave / n
__global__ void __launch_bounds__(32)
hm_final(const float* __restrict__ logits, const int* __restrict__ tgt,
         const float* __restrict__ XY, const float* __restrict__ SY,
         const float* __restrict__ sp_sum, const float* __restrict__ sg_sum,
         const float* __restrict__ ysum, float* __restrict__ out) {
  int n = blockIdx.x;
  int lane = threadIdx.x;
  const float* lg = logits + (size_t)n * KCLS;

  float mx = -3.0e38f;
  for (int k = lane; k < KCLS; k += 32) mx = fmaxf(mx, lg[k]);
  for (int o = 16; o > 0; o >>= 1) mx = fmaxf(mx, __shfl_xor(mx, o));
  float sum = 0.0f;
  for (int k = lane; k < KCLS; k += 32) sum += __expf(lg[k] - mx);
  for (int o = 16; o > 0; o >>= 1) sum += __shfl_xor(sum, o);
  float inv = 1.0f / sum;

  float spn = sp_sum[n];
  float sgn = sg_sum[n];
  const float invPn = 1.0f / (float)PN;

  for (int m = lane; m < MT; m += 32) {
    int id = tgt[m];
    id = id < 0 ? 0 : (id >= KCLS ? KCLS - 1 : id);
    float prob  = __expf(lg[id] - mx) * inv;
    float xy    = XY[(size_t)n * MT + m];
    float sy    = SY[(size_t)n * MT + m];
    float cmask = (spn - xy) * invPn;
    float cdice = 1.0f - (2.0f * sy + 1.0f) / (sgn + ysum[m] + 1.0f);
    out[(size_t)n * MT + m] = 2.0f * (-prob) + 5.0f * cmask + 5.0f * cdice;
  }
}

// ---------------------------------------------------------------- launcher
extern "C" void kernel_launch(void* const* d_in, const int* in_sizes, int n_in,
                              void* d_out, int out_size, void* d_ws, size_t ws_size,
                              hipStream_t stream) {
  const float* pred_logits = (const float*)d_in[0];   // (4,100,134) f32
  const float* pred_masks  = (const float*)d_in[1];   // (4,100,256,256) f32
  const int*   tgt_labels  = (const int*)  d_in[2];   // (80,) i32
  const float* tgt_masks   = (const float*)d_in[3];   // (80,256,256) f32
  const int*   point_idx   = (const int*)  d_in[4];   // (12544,) i32
  float* out = (float*)d_out;                         // (4,100,80) f32

  char* ws = (char*)d_ws;
  size_t off = 0;
  auto carve = [&](size_t bytes) -> char* {
    char* p = ws + off;
    off = (off + bytes + 255) & ~(size_t)255;
    return p;
  };
  _Float16* Xh = (_Float16*)carve((size_t)NQ * PN * sizeof(_Float16));
  _Float16* Sh = (_Float16*)carve((size_t)NQ * PN * sizeof(_Float16));
  _Float16* Yt = (_Float16*)carve((size_t)PN * MT * sizeof(_Float16));
  float* XYp     = (float*)carve((size_t)KSPLIT * NQ * MT * sizeof(float));
  float* SYp     = (float*)carve((size_t)KSPLIT * NQ * MT * sizeof(float));
  float* XY      = (float*)carve((size_t)NQ * MT * sizeof(float));
  float* SY      = (float*)carve((size_t)NQ * MT * sizeof(float));
  float* sp_part = (float*)carve((size_t)NQ * NCHK * sizeof(float));
  float* sg_part = (float*)carve((size_t)NQ * NCHK * sizeof(float));
  float* ys_part = (float*)carve((size_t)MT * NCHK * sizeof(float));
  float* sp_sum  = (float*)carve(NQ * sizeof(float));
  float* sg_sum  = (float*)carve(NQ * sizeof(float));
  float* ysum    = (float*)carve(MT * sizeof(float));

  hm_gather_x<<<dim3(NQ, NCHK), 256, 0, stream>>>(pred_masks, point_idx,
                                                  Xh, Sh, sp_part, sg_part);
  hm_gather_y<<<dim3(MT, NCHK), 256, 0, stream>>>(tgt_masks, point_idx,
                                                  Yt, ys_part);
  hm_reduce_rows<<<NQ + MT, 32, 0, stream>>>(sp_part, sg_part, ys_part,
                                             sp_sum, sg_sum, ysum);
  hm_wmma<<<dim3(NQ / 16, MT / 16, KSPLIT), 32, 0, stream>>>(Xh, Sh, Yt, XYp, SYp);
  hm_reduce_tiles<<<(NQ * MT) / 256, 256, 0, stream>>>(XYp, SYp, XY, SY);
  hm_final<<<NQ, 32, 0, stream>>>(pred_logits, tgt_labels, XY, SY,
                                  sp_sum, sg_sum, ysum, out);
}
